// LSTMCRF_31061203484727
// MI455X (gfx1250) — matrix operand order repaired and checked by reading, not compiled, not run on hardware
//
#include <hip/hip_runtime.h>
#include <hip/hip_bf16.h>
#include <math.h>

// ---------------------------------------------------------------------------
// BiLSTM(4 layers) + LN + Linear + CRF for MI455X (gfx1250, wave32, WMMA)
// ---------------------------------------------------------------------------

typedef __bf16 bf16_t;
typedef __attribute__((ext_vector_type(16))) __bf16 bf16x16;
typedef __attribute__((ext_vector_type(8)))  __bf16 bf16x8;
typedef __attribute__((ext_vector_type(8)))  float  f32x8;

#define B_SZ 16
#define S_SZ 256
#define E_SZ 256
#define H_SZ 256
#define L_SZ 9
#define G4H  1024     // 4*H
#define X2H  512      // 2*H
#define M_SZ (B_SZ*S_SZ)   // 4096 rows

union FragBF { bf16x16 v; bf16x8 h[2]; };

static __device__ __forceinline__ int lane_id() { return threadIdx.x & 31; }

// A fragment (16x32 bf16, M=rows): lane m = lane&15; per-lane halfwords are
// K = {base..base+7, base+16..base+23}, base = (lane<16 ? 0 : 8)  [ISA 7.12.2]
static __device__ __forceinline__ bf16x16 load_a_frag(const bf16_t* A, int lda,
                                                      int m0, int k0) {
  const int l = lane_id();
  const bf16_t* p = A + (size_t)(m0 + (l & 15)) * lda + (k0 + ((l >> 4) << 3));
  FragBF u;
  u.h[0] = *(const bf16x8*)(p);
  u.h[1] = *(const bf16x8*)(p + 16);
  return u.v;
}

// B fragment (32x16 bf16, K x N) from W stored row-major as (N x K):
// lane n = lane&15; K = {base..base+7, base+16..base+23}, base = (lane<16 ? 0 : 8)
static __device__ __forceinline__ bf16x16 load_b_frag(const bf16_t* W, int ldw,
                                                      int n0, int k0) {
  const int l = lane_id();
  const bf16_t* p = W + (size_t)(n0 + (l & 15)) * ldw + (k0 + ((l >> 4) << 3));
  FragBF u;
  u.h[0] = *(const bf16x8*)(p);
  u.h[1] = *(const bf16x8*)(p + 16);
  return u.v;
}

static __device__ __forceinline__ f32x8 wmma_bf16(bf16x16 a, bf16x16 b, f32x8 c) {
  return __builtin_amdgcn_wmma_f32_16x16x32_bf16(
      /*neg_a=*/false, a, /*neg_b=*/false, b,
      /*c_mod=*/(short)0, c, /*reuse_a=*/false, /*reuse_b=*/false);
}

static __device__ __forceinline__ float sigmoidf_(float x) {
  return 1.0f / (1.0f + __expf(-x));
}

// ---------------------------------------------------------------------------
// Small helper kernels
// ---------------------------------------------------------------------------

// Embedding gather -> bf16 activation buffer (cols 0..255 of stride-512 rows)
__global__ void embed_kernel(const int* __restrict__ ids,
                             const float* __restrict__ emb,
                             bf16_t* __restrict__ xbf) {
  const int i = blockIdx.x * blockDim.x + threadIdx.x;   // B*S*E threads
  const int bs = i >> 8;
  const int cc = i & 255;
  const int id = ids[bs];
  xbf[(size_t)bs * X2H + cc] = (bf16_t)emb[(size_t)id * E_SZ + cc];
}

__global__ void convert_kernel(const float* __restrict__ in,
                               bf16_t* __restrict__ out, int n) {
  const int i = blockIdx.x * blockDim.x + threadIdx.x;
  if (i < n) out[i] = (bf16_t)in[i];
}

// lin_w (9 x 512) -> zero-padded (16 x 512) bf16
__global__ void pad_linw_kernel(const float* __restrict__ w,
                                bf16_t* __restrict__ wpad) {
  const int i = blockIdx.x * blockDim.x + threadIdx.x;   // 16*512 threads
  const int r = i >> 9, cc = i & 511;
  wpad[i] = (bf16_t)((r < L_SZ) ? w[r * X2H + cc] : 0.0f);
}

// ---------------------------------------------------------------------------
// Input projection: xg[dir][t][b][gate] = Xbf(M x K) @ W_ih(dir)^T + b_ih+b_hh
// One wave computes a 16x64 output block (1 A-frag amortized over 4 B-frags)
// to cut L2 traffic 4x on the A side. Output is time-major for the scan.
// ---------------------------------------------------------------------------
__global__ __launch_bounds__(256) void gemm_xg_kernel(
    const bf16_t* __restrict__ Xbf,    // [M, 512] (only first K cols used)
    const bf16_t* __restrict__ Wih,    // [2][1024 * K]
    const float*  __restrict__ bias,   // [2][2][1024] for this layer
    float* __restrict__ xg,            // [2][S*B*1024]
    int K) {
  const int dir  = blockIdx.y;
  const int tile = blockIdx.x * 8 + (threadIdx.x >> 5);  // 256 * 16 tiles
  const int m0 = (tile >> 4) * 16;
  const int n0 = (tile & 15) * 64;
  const bf16_t* W = Wih + (size_t)dir * G4H * K;

  const f32x8 zero = {0.f,0.f,0.f,0.f,0.f,0.f,0.f,0.f};
  f32x8 acc[4];
#pragma unroll
  for (int q = 0; q < 4; ++q) acc[q] = zero;

  for (int k0 = 0; k0 < K; k0 += 32) {
    bf16x16 a = load_a_frag(Xbf, X2H, m0, k0);
#pragma unroll
    for (int q = 0; q < 4; ++q) {
      bf16x16 b = load_b_frag(W, K, n0 + q * 16, k0);
      acc[q] = wmma_bf16(a, b, acc[q]);
    }
  }

  const int l = lane_id();
  float* out = xg + (size_t)dir * ((size_t)S_SZ * B_SZ * G4H);
#pragma unroll
  for (int q = 0; q < 4; ++q) {
    const int n = n0 + q * 16 + (l & 15);
    const float bs = bias[dir * 2048 + n] + bias[dir * 2048 + 1024 + n];
#pragma unroll
    for (int j = 0; j < 8; ++j) {
      const int m = m0 + j + ((l >> 4) << 3);
      const int b = m >> 8;     // m / S
      const int s = m & 255;    // m % S
      out[((size_t)(s * B_SZ + b)) * G4H + n] = acc[q][j] + bs;  // time-major
    }
  }
}

// ---------------------------------------------------------------------------
// Recurrent scan: one workgroup (16 waves) per direction, whole sequence.
// Wave w owns h-columns [16w,16w+16) for all 4 gates; w_hh fragments are
// preloaded into 256 VGPRs. h lives in LDS (bf16, padded rows), c in regs.
// xg for step t+1 is prefetched into registers during step t's WMMA chain
// so the L2 round-trip is hidden behind matrix math on the serial path.
// ---------------------------------------------------------------------------
__global__ __launch_bounds__(512) void lstm_scan_kernel(
    const float*  __restrict__ xg,     // [2][S*B*1024] (biases included)
    const bf16_t* __restrict__ whh,    // [2][1024 * 256]
    float* __restrict__ xout) {        // [B,S,512]; dir d -> cols d*256..
  const int dir = blockIdx.x;
  const float*  xgd = xg  + (size_t)dir * ((size_t)S_SZ * B_SZ * G4H);
  const bf16_t* W   = whh + (size_t)dir * ((size_t)G4H * H_SZ);

  __shared__ __align__(16) bf16_t hbuf[B_SZ][H_SZ + 8];  // pad breaks conflicts

  const int wave = threadIdx.x >> 5;   // 0..15, h-tile index
  const int l    = threadIdx.x & 31;
  const int row  = (l >> 4) << 3;      // +0 / +8 (C-layout M offset)
  const int col  = l & 15;             // C-layout N / A-layout M

  int n0[4];
#pragma unroll
  for (int g = 0; g < 4; ++g) n0[g] = g * H_SZ + wave * 16;

  // Resident w_hh slice: 4 gates x 8 K-fragments = 256 VGPRs of bf16
  bf16x16 Bf[4][8];
#pragma unroll
  for (int g = 0; g < 4; ++g)
#pragma unroll
    for (int kf = 0; kf < 8; ++kf)
      Bf[g][kf] = load_b_frag(W, H_SZ, n0[g], kf * 32);

  for (int i = threadIdx.x; i < B_SZ * (H_SZ + 8); i += blockDim.x)
    ((bf16_t*)hbuf)[i] = (bf16_t)0.0f;

  f32x8 c = {0.f,0.f,0.f,0.f,0.f,0.f,0.f,0.f};

  // Prime the pipeline: xg slice for the first step
  f32x8 nxt[4];
  {
    const int tt0 = dir ? (S_SZ - 1) : 0;
    const float* xgt = xgd + (size_t)tt0 * (B_SZ * G4H);
#pragma unroll
    for (int g = 0; g < 4; ++g)
#pragma unroll
      for (int j = 0; j < 8; ++j)
        nxt[g][j] = xgt[(j + row) * G4H + n0[g] + col];
  }

  __syncthreads();

  for (int t = 0; t < S_SZ; ++t) {
    const int tt = dir ? (S_SZ - 1 - t) : t;

    f32x8 acc[4];
#pragma unroll
    for (int g = 0; g < 4; ++g) acc[g] = nxt[g];

    // Prefetch next step's xg (clamped at the tail; redundant but uniform)
    const int tn  = (t + 1 < S_SZ) ? (t + 1) : t;
    const int ttn = dir ? (S_SZ - 1 - tn) : tn;
    const float* xgn = xgd + (size_t)ttn * (B_SZ * G4H);
#pragma unroll
    for (int g = 0; g < 4; ++g)
#pragma unroll
      for (int j = 0; j < 8; ++j)
        nxt[g][j] = xgn[(j + row) * G4H + n0[g] + col];

#pragma unroll
    for (int kf = 0; kf < 8; ++kf) {
      // A fragment of h from LDS: m = col, K chunks at kf*32+row and +16
      FragBF u;
      const bf16_t* p = &hbuf[col][kf * 32 + row];
      u.h[0] = *(const bf16x8*)(p);
      u.h[1] = *(const bf16x8*)(p + 16);
#pragma unroll
      for (int g = 0; g < 4; ++g)
        acc[g] = wmma_bf16(u.v, Bf[g][kf], acc[g]);
    }

    __syncthreads();   // all hbuf reads done before overwrite

#pragma unroll
    for (int j = 0; j < 8; ++j) {
      const int b  = j + row;
      const int hc = wave * 16 + col;
      const float iv = sigmoidf_(acc[0][j]);
      const float fv = sigmoidf_(acc[1][j]);
      const float gv = tanhf(acc[2][j]);
      const float ov = sigmoidf_(acc[3][j]);
      const float cv = fv * c[j] + iv * gv;
      c[j] = cv;
      const float hv = ov * tanhf(cv);
      hbuf[b][hc] = (bf16_t)hv;
      xout[((size_t)b * S_SZ + tt) * X2H + dir * H_SZ + hc] = hv;
    }

    __syncthreads();   // hbuf updated for next step
  }
}

// ---------------------------------------------------------------------------
// LayerNorm: one wave32 per row of 512, shuffle reduction, bf16 output
// ---------------------------------------------------------------------------
__global__ __launch_bounds__(256) void ln_kernel(const float* __restrict__ x,
                                                 const float* __restrict__ g,
                                                 const float* __restrict__ bta,
                                                 bf16_t* __restrict__ xhat) {
  const int w = blockIdx.x * (blockDim.x >> 5) + (threadIdx.x >> 5);
  const int l = threadIdx.x & 31;
  const float* row = x + (size_t)w * X2H;
  float s = 0.f, ss = 0.f;
  for (int i = l; i < X2H; i += 32) { const float v = row[i]; s += v; ss += v * v; }
#pragma unroll
  for (int off = 16; off; off >>= 1) {
    s  += __shfl_xor(s,  off, 32);
    ss += __shfl_xor(ss, off, 32);
  }
  const float mu = s * (1.0f / X2H);
  const float r  = rsqrtf(ss * (1.0f / X2H) - mu * mu + 1e-5f);
  bf16_t* o = xhat + (size_t)w * X2H;
  for (int i = l; i < X2H; i += 32)
    o[i] = (bf16_t)((row[i] - mu) * r * g[i] + bta[i]);
}

// ---------------------------------------------------------------------------
// Logits: M x 16 (padded from 9) x K=512 via WMMA; one wave per M-tile
// ---------------------------------------------------------------------------
__global__ __launch_bounds__(256) void logits_kernel(
    const bf16_t* __restrict__ xhat, const bf16_t* __restrict__ wpad,
    const float* __restrict__ lin_b, float* __restrict__ out) {
  const int tile = blockIdx.x * 8 + (threadIdx.x >> 5);  // 0..255
  const int m0 = tile * 16;
  f32x8 acc = {0.f,0.f,0.f,0.f,0.f,0.f,0.f,0.f};
#pragma unroll
  for (int k0 = 0; k0 < X2H; k0 += 32)
    acc = wmma_bf16(load_a_frag(xhat, X2H, m0, k0),
                    load_b_frag(wpad, X2H, 0, k0), acc);
  const int l = lane_id();
  const int n = l & 15;
  if (n < L_SZ) {
    const float bb = lin_b[n];
#pragma unroll
    for (int j = 0; j < 8; ++j) {
      const int m = m0 + j + ((l >> 4) << 3);
      out[(size_t)m * L_SZ + n] = acc[j] + bb;
    }
  }
}

// ---------------------------------------------------------------------------
// CRF numerator + forward-algorithm denominator; loss -> out[B*S*L]
// ---------------------------------------------------------------------------
__global__ __launch_bounds__(256) void crf_kernel(
    const float* __restrict__ logits, const int* __restrict__ mask,
    const int* __restrict__ labels, const float* __restrict__ cstart,
    const float* __restrict__ cend, const float* __restrict__ ctrans,
    float* __restrict__ loss_out) {
  __shared__ float alpha[2][B_SZ][12];
  __shared__ float trans[L_SZ][12];
  __shared__ float numv[B_SZ], denv[B_SZ];
  const int tid = threadIdx.x;
  if (tid < L_SZ * L_SZ) trans[tid / L_SZ][tid % L_SZ] = ctrans[tid];
  if (tid < 144) {
    const int b = tid / L_SZ, j = tid % L_SZ;
    alpha[0][b][j] = cstart[j] + logits[((size_t)b * S_SZ) * L_SZ + j];
  }
  if (tid >= 144 && tid < 160) {
    const int b = tid - 144;
    const int f = labels[b * S_SZ];
    numv[b] = cstart[f] + logits[((size_t)b * S_SZ) * L_SZ + f];
  }
  __syncthreads();
  int cur = 0;
  for (int t = 1; t < S_SZ; ++t) {
    if (tid < 144) {
      const int b = tid / L_SZ, j = tid % L_SZ;
      const float m = (float)mask[b * S_SZ + t];
      const float e = logits[((size_t)b * S_SZ + t) * L_SZ + j];
      float mx = -3.0e38f;
#pragma unroll
      for (int i = 0; i < L_SZ; ++i) mx = fmaxf(mx, alpha[cur][b][i] + trans[i][j]);
      float sum = 0.f;
#pragma unroll
      for (int i = 0; i < L_SZ; ++i) sum += __expf(alpha[cur][b][i] + trans[i][j] - mx);
      const float nxt = mx + __logf(sum) + e;
      alpha[1 - cur][b][j] = (m > 0.f) ? nxt : alpha[cur][b][j];
    }
    if (tid >= 144 && tid < 160) {
      const int b = tid - 144;
      const float m = (float)mask[b * S_SZ + t];
      const int tp = labels[b * S_SZ + t - 1], tc = labels[b * S_SZ + t];
      numv[b] += (trans[tp][tc] + logits[((size_t)b * S_SZ + t) * L_SZ + tc]) * m;
    }
    __syncthreads();
    cur ^= 1;
  }
  if (tid < B_SZ) {
    const int b = tid;
    int lsum = 0;
    for (int t = 0; t < S_SZ; ++t) lsum += mask[b * S_SZ + t];
    numv[b] += cend[labels[b * S_SZ + (lsum - 1)]];
    float mx = -3.0e38f;
#pragma unroll
    for (int j = 0; j < L_SZ; ++j) mx = fmaxf(mx, alpha[cur][b][j] + cend[j]);
    float s = 0.f;
#pragma unroll
    for (int j = 0; j < L_SZ; ++j) s += __expf(alpha[cur][b][j] + cend[j] - mx);
    denv[b] = mx + __logf(s);
  }
  __syncthreads();
  if (tid == 0) {
    float lz = 0.f;
    for (int b = 0; b < B_SZ; ++b) lz += numv[b] - denv[b];
    loss_out[0] = -lz;
  }
}

// ---------------------------------------------------------------------------
extern "C" void kernel_launch(void* const* d_in, const int* in_sizes, int n_in,
                              void* d_out, int out_size, void* d_ws, size_t ws_size,
                              hipStream_t stream) {
  (void)in_sizes; (void)n_in; (void)out_size; (void)ws_size;
  const int*   input_ids = (const int*)  d_in[0];
  const int*   attn_mask = (const int*)  d_in[1];
  const int*   labels    = (const int*)  d_in[2];
  const float* embedding = (const float*)d_in[3];
  const float* w_ih_l0   = (const float*)d_in[4];   // (2,1024,256)
  const float* w_ih_rest = (const float*)d_in[5];   // (3,2,1024,512)
  const float* w_hh      = (const float*)d_in[6];   // (4,2,1024,256)
  const float* bvec      = (const float*)d_in[7];   // (4,2,2,1024)
  const float* ln_g      = (const float*)d_in[8];
  const float* ln_b      = (const float*)d_in[9];
  const float* lin_w     = (const float*)d_in[10];  // (9,512)
  const float* lin_b     = (const float*)d_in[11];
  const float* crf_start = (const float*)d_in[12];
  const float* crf_end   = (const float*)d_in[13];
  const float* crf_trans = (const float*)d_in[14];

  uint8_t* ws = (uint8_t*)d_ws;
  size_t off = 0;
  auto walloc = [&](size_t bytes) -> void* {
    off = (off + 255) & ~(size_t)255;
    void* p = ws + off;
    off += bytes;
    return p;
  };
  float*  xA      = (float*) walloc((size_t)M_SZ * X2H * sizeof(float));      // 8 MB
  bf16_t* xAbf    = (bf16_t*)walloc((size_t)M_SZ * X2H * sizeof(bf16_t));     // 4 MB
  float*  xg      = (float*) walloc((size_t)2 * S_SZ * B_SZ * G4H * 4);       // 32 MB
  bf16_t* wihbf   = (bf16_t*)walloc((size_t)2 * G4H * X2H * sizeof(bf16_t));  // 2 MB
  bf16_t* whhbf   = (bf16_t*)walloc((size_t)2 * G4H * H_SZ * sizeof(bf16_t)); // 1 MB
  bf16_t* linwpad = (bf16_t*)walloc((size_t)16 * X2H * sizeof(bf16_t));

  embed_kernel<<<(B_SZ * S_SZ * E_SZ) / 256, 256, 0, stream>>>(input_ids, embedding, xAbf);
  pad_linw_kernel<<<(16 * X2H) / 256, 256, 0, stream>>>(lin_w, linwpad);

  for (int l = 0; l < 4; ++l) {
    const int K = (l == 0) ? E_SZ : X2H;
    const float* wih = (l == 0) ? w_ih_l0
                                : (w_ih_rest + (size_t)(l - 1) * 2 * G4H * X2H);
    const int nWih = 2 * G4H * K;
    convert_kernel<<<(nWih + 255) / 256, 256, 0, stream>>>(wih, wihbf, nWih);
    const int nWhh = 2 * G4H * H_SZ;
    convert_kernel<<<(nWhh + 255) / 256, 256, 0, stream>>>(
        w_hh + (size_t)l * nWhh, whhbf, nWhh);

    // 256 M-tiles x 16 N-groups (16x64 per wave), 8 waves/block, 2 dirs
    gemm_xg_kernel<<<dim3((M_SZ / 16) * (G4H / 64) / 8, 2), 256, 0, stream>>>(
        xAbf, wihbf, bvec + (size_t)l * 4096, xg, K);

    lstm_scan_kernel<<<2, 512, 0, stream>>>(xg, whhbf, xA);

    if (l < 3) {
      const int nAct = M_SZ * X2H;
      convert_kernel<<<(nAct + 255) / 256, 256, 0, stream>>>(xA, xAbf, nAct);
    }
  }

  ln_kernel<<<M_SZ / 8, 256, 0, stream>>>(xA, ln_g, ln_b, xAbf);

  float* logits = (float*)d_out;
  logits_kernel<<<(M_SZ / 16) / 8, 256, 0, stream>>>(xAbf, linwpad, lin_b, logits);

  crf_kernel<<<1, 256, 0, stream>>>(logits, attn_mask, labels, crf_start,
                                    crf_end, crf_trans,
                                    logits + (size_t)B_SZ * S_SZ * L_SZ);
}
